// DNC_80444737454115
// MI455X (gfx1250) — matrix operand order repaired and checked by reading, not compiled, not run on hardware
//
#include <hip/hip_runtime.h>
#include <hip/hip_bf16.h>
#include <math.h>

// ---------------- problem constants ----------------
#define B_   8
#define T_   32
#define IN_  256
#define H_   512
#define H4_  2048
#define N_   1024
#define CW_  64
#define R_   4
#define RV_  256
#define XI_  471
#define XIP_ 480     // xi padded to 30 WMMA N-tiles
#define CLIPV 20.0f
#define EPS_ 1e-6f

typedef __attribute__((ext_vector_type(16))) _Float16 v16h;
typedef __attribute__((ext_vector_type(8)))  float    v8f;

__device__ __forceinline__ float sigm(float x) { return 1.0f / (1.0f + expf(-x)); }
__device__ __forceinline__ float sp_f(float x) { return (x > 20.0f) ? x : log1pf(expf(x)); }

// ---------------- WMMA fragment helpers (CDNA5 16x16x32 f16) ----------------
// A fragment (16x32, f16) from LDS row-major sA[16][1024].
// lane L: row m = L%16 ; lanes<16 hold K {0..7,16..23}, lanes>=16 add 8.
__device__ __forceinline__ v16h load_a_frag(const _Float16* sA, int lane, int k0) {
  v16h a;
  int m = lane & 15;
  int sel = (lane & 16) ? 8 : 0;
  const _Float16* row = sA + m * 1024;
#pragma unroll
  for (int j = 0; j < 8; ++j) {
    int bk = k0 + sel + ((j < 4) ? (2 * j) : (16 + 2 * (j - 4)));
    a[2 * j]     = row[bk];
    a[2 * j + 1] = row[bk + 1];
  }
  return a;
}

// B fragment from a pre-packed global buffer: element order is exactly the
// per-lane fragment layout, so each lane does one contiguous 32B load.
__device__ __forceinline__ const _Float16* b_frag_ptr(const _Float16* wp, int KT, int ntile, int kt, int lane) {
  return wp + (((((size_t)ntile) * KT + kt) * 32 + lane) << 4);
}
__device__ __forceinline__ v16h load_b_frag(const _Float16* wp, int KT, int ntile, int kt, int lane) {
  return *(const v16h*)b_frag_ptr(wp, KT, ntile, kt, lane);
}

__device__ __forceinline__ v8f wmma16(v16h a, v16h b, v8f c) {
  return __builtin_amdgcn_wmma_f32_16x16x32_f16(false, a, false, b, (short)0, c, false, false);
}

// ---------------- weight packing: f32 [n][k] -> f16 WMMA-B fragment order ----
__global__ void pack_w_kernel(_Float16* dst, const float* a, const float* b,
                              int KT, int rows_valid, int total) {
  int id = blockIdx.x * 256 + threadIdx.x;
  if (id >= total) return;
  int i  = id & 15;
  int L  = (id >> 4) & 31;
  int kt = (id >> 9) % KT;
  int nt = id / (KT * 512);
  int n  = nt * 16 + (L & 15);
  int k  = kt * 32 + ((L & 16) ? 8 : 0) + ((i & 8) ? 16 : 0) + (i & 7);
  float v = 0.0f;
  if (n < rows_valid) {
    if (k < 512)      v = a[n * 512 + k];
    else if (b)       v = b[n * 512 + (k - 512)];
  }
  dst[id] = (_Float16)v;
}

__global__ void fill_kernel(float* p, float v, int n) {
  int id = blockIdx.x * 256 + threadIdx.x;
  if (id < n) p[id] = v;
}

__global__ void init_h_kernel(const float* h0, float* hA, float* cA, float* hB, float* cB) {
  int id = blockIdx.x * 256 + threadIdx.x;
  if (id < B_ * H_) {
    float a = h0[id], b = h0[B_ * H_ + id];
    hA[id] = a; cA[id] = a; hB[id] = b; cB[id] = b;
  }
}

// ---------------- fused LSTM0 + LSTM1 + xi kernel (one workgroup, WMMA) -----
// launch_bounds(256, 1): single resident workgroup -> maximum VGPR budget so
// the 16 staged B fragments keep distinct registers (batched load clauses).
__global__ __launch_bounds__(256, 1) void lstm_step_kernel(
    const float* __restrict__ x,
    const float* __restrict__ bih0, const float* __restrict__ bhh0,
    const float* __restrict__ bih1, const float* __restrict__ bhh1,
    const float* __restrict__ bif,
    const _Float16* __restrict__ w0h, const _Float16* __restrict__ w1h,
    const _Float16* __restrict__ wifh,
    float* hA, float* cA, float* hB, float* cB, float* outg, float* xig, int t) {
  __shared__ _Float16 sA[16 * 1024];   // 32 KB A staging (rows 8..15 stay zero)
  int tid  = threadIdx.x;
  int lane = tid & 31;
  int wave = tid >> 5;
  int j0   = wave * 64;               // each wave owns gate columns [j0, j0+64)

  // Stage A0 = [x_t | zeros(last_read, per reference) | hA_old]
  for (int idx = tid; idx < 16 * 1024; idx += 256) {
    int m = idx >> 10, k = idx & 1023;
    float v = 0.0f;
    if (m < 8) {
      if (k < 256)        v = x[(m * T_ + t) * IN_ + k];
      else if (k >= 512)  v = hA[m * H_ + (k - 512)];
    }
    sA[idx] = (_Float16)v;
  }
  // Warm the WGP cache for the first K-steps of layer-0 weights while staging.
#pragma unroll
  for (int g = 0; g < 4; ++g)
#pragma unroll
    for (int tt = 0; tt < 4; ++tt)
      __builtin_prefetch(b_frag_ptr(w0h, 32, (g * 512 + j0 + tt * 16) >> 4, 0, lane), 0, 1);
  __syncthreads();

  v8f acc[16];
  const v8f vz = {0, 0, 0, 0, 0, 0, 0, 0};

  // ---- layer 0: gates = [x|0|hA] @ [w_ih0|w_hh0]^T
  // Per K-step: prefetch next K-step's fragments, batch-issue this K-step's
  // 16 B-fragment loads, then 16 independent WMMAs.
#pragma unroll
  for (int i = 0; i < 16; ++i) acc[i] = vz;
  for (int kt = 0; kt < 32; ++kt) {
    v16h a = load_a_frag(sA, lane, kt * 32);
    if (kt + 1 < 32) {
#pragma unroll
      for (int g = 0; g < 4; ++g)
#pragma unroll
        for (int tt = 0; tt < 4; ++tt)
          __builtin_prefetch(b_frag_ptr(w0h, 32, (g * 512 + j0 + tt * 16) >> 4, kt + 1, lane), 0, 1);
    }
    v16h bf[16];
#pragma unroll
    for (int g = 0; g < 4; ++g)
#pragma unroll
      for (int tt = 0; tt < 4; ++tt)
        bf[g * 4 + tt] = load_b_frag(w0h, 32, (g * 512 + j0 + tt * 16) >> 4, kt, lane);
#pragma unroll
    for (int i = 0; i < 16; ++i)
      acc[i] = wmma16(a, bf[i], acc[i]);
  }
  __syncthreads();   // all waves done reading sA before we overwrite it

  if (lane < 16) {
#pragma unroll
    for (int tt = 0; tt < 4; ++tt) {
      int j = j0 + tt * 16 + lane;
      float bi = bih0[j] + bhh0[j];
      float bf = bih0[512 + j] + bhh0[512 + j];
      float bg = bih0[1024 + j] + bhh0[1024 + j];
      float bo = bih0[1536 + j] + bhh0[1536 + j];
#pragma unroll
      for (int m = 0; m < 8; ++m) {
        float gi = acc[0 * 4 + tt][m] + bi;
        float gf = acc[1 * 4 + tt][m] + bf;
        float gg = acc[2 * 4 + tt][m] + bg;
        float go = acc[3 * 4 + tt][m] + bo;
        float cn = sigm(gf) * cA[m * H_ + j] + sigm(gi) * tanhf(gg);
        float hn = sigm(go) * tanhf(cn);
        cA[m * H_ + j] = cn; hA[m * H_ + j] = hn;
        sA[m * 1024 + j] = (_Float16)hn;          // A1 low half = hA_new
      }
    }
  }
  // A1 high half = hB_old (cooperative)
  for (int idx = tid; idx < 8 * 512; idx += 256) {
    int m = idx >> 9, k = idx & 511;
    sA[m * 1024 + 512 + k] = (_Float16)hB[m * H_ + k];
  }
  __syncthreads();

  // ---- layer 1: gates = [hA_new|hB_old] @ [w_ih1|w_hh1]^T
#pragma unroll
  for (int i = 0; i < 16; ++i) acc[i] = vz;
  for (int kt = 0; kt < 32; ++kt) {
    v16h a = load_a_frag(sA, lane, kt * 32);
    if (kt + 1 < 32) {
#pragma unroll
      for (int g = 0; g < 4; ++g)
#pragma unroll
        for (int tt = 0; tt < 4; ++tt)
          __builtin_prefetch(b_frag_ptr(w1h, 32, (g * 512 + j0 + tt * 16) >> 4, kt + 1, lane), 0, 1);
    }
    v16h bf[16];
#pragma unroll
    for (int g = 0; g < 4; ++g)
#pragma unroll
      for (int tt = 0; tt < 4; ++tt)
        bf[g * 4 + tt] = load_b_frag(w1h, 32, (g * 512 + j0 + tt * 16) >> 4, kt, lane);
#pragma unroll
    for (int i = 0; i < 16; ++i)
      acc[i] = wmma16(a, bf[i], acc[i]);
  }
  __syncthreads();

  if (lane < 16) {
#pragma unroll
    for (int tt = 0; tt < 4; ++tt) {
      int j = j0 + tt * 16 + lane;
      float bi = bih1[j] + bhh1[j];
      float bf = bih1[512 + j] + bhh1[512 + j];
      float bg = bih1[1024 + j] + bhh1[1024 + j];
      float bo = bih1[1536 + j] + bhh1[1536 + j];
#pragma unroll
      for (int m = 0; m < 8; ++m) {
        float gi = acc[0 * 4 + tt][m] + bi;
        float gf = acc[1 * 4 + tt][m] + bf;
        float gg = acc[2 * 4 + tt][m] + bg;
        float go = acc[3 * 4 + tt][m] + bo;
        float cn = sigm(gf) * cB[m * H_ + j] + sigm(gi) * tanhf(gg);
        float hn = sigm(go) * tanhf(cn);
        cB[m * H_ + j] = cn; hB[m * H_ + j] = hn;
        float oc = fminf(CLIPV, fmaxf(-CLIPV, hn));
        outg[m * H_ + j] = oc;
        sA[m * 1024 + j] = (_Float16)oc;          // A for xi GEMM (K=512)
      }
    }
  }
  __syncthreads();

  // ---- xi = out @ w_if^T + b_if  (30 N-tiles over 8 waves, 4 accs/wave)
  {
    v8f xacc[4] = {vz, vz, vz, vz};
    int tiles[4];
#pragma unroll
    for (int p = 0; p < 4; ++p) {
      int tl = wave + 8 * p;
      tiles[p] = (tl < 30) ? tl : 0;   // redundant compute keeps EXEC uniform
    }
    for (int kt = 0; kt < 16; ++kt) {
      v16h a = load_a_frag(sA, lane, kt * 32);
      if (kt + 1 < 16) {
#pragma unroll
        for (int p = 0; p < 4; ++p)
          __builtin_prefetch(b_frag_ptr(wifh, 16, tiles[p], kt + 1, lane), 0, 1);
      }
      v16h bf[4];
#pragma unroll
      for (int p = 0; p < 4; ++p) bf[p] = load_b_frag(wifh, 16, tiles[p], kt, lane);
#pragma unroll
      for (int p = 0; p < 4; ++p) xacc[p] = wmma16(a, bf[p], xacc[p]);
    }
    if (lane < 16) {
#pragma unroll
      for (int p = 0; p < 4; ++p) {
        int tl = wave + 8 * p;
        if (tl < 30) {
          int n = tl * 16 + lane;
          if (n < XI_) {
            float bias = bif[n];
#pragma unroll
            for (int m = 0; m < 8; ++m) xig[m * XIP_ + n] = xacc[p][m] + bias;
          }
        }
      }
    }
  }
}

// ---------------- content-based write weighting + allocation (per batch) ----
__global__ __launch_bounds__(1024) void content_kernel(
    const float* __restrict__ xi_all, const float* __restrict__ mem,
    float* rdkeys, float* rdstr, float* eraseg, float* wvecg, float* modesg,
    float* ww, const float* __restrict__ rw, float* usage, float* sumwv) {
  __shared__ float s_red[1024];
  __shared__ float s_key[1024];
  __shared__ int   s_idx[1024];
  __shared__ float s_val[1024];
  __shared__ float s_wkey[64];
  __shared__ float s_free[4];
  __shared__ float s_sc[3];   // 0: write_str, 1: alloc_gate, 2: write_gate
  int b = blockIdx.x, tid = threadIdx.x, n = tid;
  const float* xb = xi_all + b * XIP_;

  if (tid < 256)        rdkeys[b * RV_ + tid] = tanhf(xb[tid]);
  else if (tid < 260)   rdstr[b * R_ + (tid - 256)] = sp_f(xb[tid]);
  else if (tid < 324)   s_wkey[tid - 260] = tanhf(xb[tid]);
  else if (tid == 324)  s_sc[0] = sp_f(xb[324]);
  else if (tid < 389)   eraseg[b * CW_ + (tid - 325)] = sigm(xb[tid]);
  else if (tid < 453)   wvecg[b * CW_ + (tid - 389)] = tanhf(xb[tid]);
  else if (tid < 457)   s_free[tid - 453] = sigm(xb[tid]);
  else if (tid == 457)  s_sc[1] = sigm(xb[457]);
  else if (tid == 458)  s_sc[2] = sigm(xb[458]);
  else if (tid < 463) {
    int r = tid - 459;
    float a0 = xb[459 + 3 * r], a1 = xb[460 + 3 * r], a2 = xb[461 + 3 * r];
    float mx = fmaxf(a0, fmaxf(a1, a2));
    float e0 = expf(a0 - mx), e1 = expf(a1 - mx), e2 = expf(a2 - mx);
    float s = e0 + e1 + e2;
    modesg[(b * R_ + r) * 3 + 0] = e0 / s;
    modesg[(b * R_ + r) * 3 + 1] = e1 / s;
    modesg[(b * R_ + r) * 3 + 2] = e2 / s;
  }
  __syncthreads();

  // usage update (uses previous-step ww, rw)
  float ww_o = ww[b * N_ + n];
  float us = usage[b * N_ + n];
  us = us + (1.0f - us) * ww_o;
  float pf = 1.0f;
#pragma unroll
  for (int r = 0; r < 4; ++r) pf *= (1.0f - s_free[r] * rw[(b * R_ + r) * N_ + n]);
  us *= pf;
  usage[b * N_ + n] = us;
  float u = EPS_ + (1.0f - EPS_) * us;

  // write content weighting (cosine vs old memory, softmax over N)
  float norm = 0, dot = 0, kn = 0;
  const float* mrow = mem + ((size_t)(b * N_ + n)) * CW_;
  for (int w = 0; w < CW_; ++w) {
    float mv = mrow[w], kv = s_wkey[w];
    norm += mv * mv; dot += mv * kv; kn += kv * kv;
  }
  float score = dot / ((sqrtf(norm) + EPS_) * (sqrtf(kn) + EPS_)) * s_sc[0];
  s_red[tid] = score; __syncthreads();
  for (int s = 512; s > 0; s >>= 1) { if (tid < s) s_red[tid] = fmaxf(s_red[tid], s_red[tid + s]); __syncthreads(); }
  float mx = s_red[0]; __syncthreads();
  float e = expf(score - mx);
  s_red[tid] = e; __syncthreads();
  for (int s = 512; s > 0; s >>= 1) { if (tid < s) s_red[tid] += s_red[tid + s]; __syncthreads(); }
  float wcw = e / s_red[0];
  __syncthreads();

  // allocation: stable ascending sort of u (bitonic on (u, idx))
  s_key[tid] = u; s_idx[tid] = tid;
  __syncthreads();
  for (int k = 2; k <= 1024; k <<= 1)
    for (int j = k >> 1; j > 0; j >>= 1) {
      int ixj = tid ^ j;
      if (ixj > tid) {
        bool up = ((tid & k) == 0);
        float ka = s_key[tid], kb = s_key[ixj];
        int ia = s_idx[tid], ib = s_idx[ixj];
        bool gt = (ka > kb) || (ka == kb && ia > ib);
        if (gt == up) { s_key[tid] = kb; s_key[ixj] = ka; s_idx[tid] = ib; s_idx[ixj] = ia; }
      }
      __syncthreads();
    }
  // exclusive prefix product of sorted u
  s_val[tid] = s_key[tid];
  __syncthreads();
  for (int off = 1; off < 1024; off <<= 1) {
    float o = (tid >= off) ? s_val[tid - off] : 1.0f;
    __syncthreads();
    s_val[tid] *= o;
    __syncthreads();
  }
  float excl = (tid == 0) ? 1.0f : s_val[tid - 1];
  float asort = (1.0f - s_key[tid]) * excl;
  s_red[s_idx[tid]] = asort;       // scatter back to memory order
  __syncthreads();
  float alloc = s_red[tid];

  float ag = s_sc[1], wg = s_sc[2];
  float wwn = wg * (ag * alloc + (1.0f - ag) * wcw);
  ww[b * N_ + n] = wwn;
  __syncthreads();
  s_key[tid] = wwn; __syncthreads();
  for (int s = 512; s > 0; s >>= 1) { if (tid < s) s_key[tid] += s_key[tid + s]; __syncthreads(); }
  if (tid == 0) sumwv[b] = s_key[0];
}

// ---------------- link update + forward weights (row-parallel) --------------
__global__ __launch_bounds__(256) void link_fw_kernel(
    float* link, const float* __restrict__ ww, const float* __restrict__ prec,
    const float* __restrict__ rw, float* fw) {
  __shared__ float s_red[256];
  int blk = blockIdx.x;
  int b = blk >> 10, i = blk & (N_ - 1);
  int tid = threadIdx.x;
  float wv_i = ww[b * N_ + i];
  size_t base = ((((size_t)b) << 10) + i) << 10;    // b*N*N + i*N
  float fp[4] = {0, 0, 0, 0};
  for (int j = tid; j < N_; j += 256) {
    float wv_j = ww[b * N_ + j];
    float pj = prec[b * N_ + j];
    float L = link[base + j];
    float Ln = (i == j) ? 0.0f : (1.0f - wv_i - wv_j) * L + wv_i * pj;
    link[base + j] = Ln;
#pragma unroll
    for (int r = 0; r < 4; ++r) fp[r] += Ln * rw[(b * R_ + r) * N_ + j];
  }
  for (int r = 0; r < 4; ++r) {
    s_red[tid] = fp[r]; __syncthreads();
    for (int s = 128; s > 0; s >>= 1) { if (tid < s) s_red[tid] += s_red[tid + s]; __syncthreads(); }
    if (tid == 0) fw[(b * R_ + r) * N_ + i] = s_red[0];
    __syncthreads();
  }
}

// ---------------- backward weights (column-parallel, coalesced) -------------
__global__ __launch_bounds__(1024) void bw_kernel(
    const float* __restrict__ link, const float* __restrict__ rw, float* bw) {
  int b = blockIdx.x, j = threadIdx.x;
  float a0 = 0, a1 = 0, a2 = 0, a3 = 0;
  size_t base = (((size_t)b) << 20) + j;
  for (int i = 0; i < N_; ++i) {
    float L = link[base + (((size_t)i) << 10)];
    a0 += rw[(b * R_ + 0) * N_ + i] * L;
    a1 += rw[(b * R_ + 1) * N_ + i] * L;
    a2 += rw[(b * R_ + 2) * N_ + i] * L;
    a3 += rw[(b * R_ + 3) * N_ + i] * L;
  }
  bw[(b * R_ + 0) * N_ + j] = a0;
  bw[(b * R_ + 1) * N_ + j] = a1;
  bw[(b * R_ + 2) * N_ + j] = a2;
  bw[(b * R_ + 3) * N_ + j] = a3;
}

// ---------------- memory write + precedence update --------------------------
__global__ void mem_write_kernel(float* mem, const float* __restrict__ ww,
                                 const float* __restrict__ eraseg,
                                 const float* __restrict__ wvecg,
                                 float* prec, const float* __restrict__ sumwv) {
  int id = blockIdx.x * 256 + threadIdx.x;      // < B*N*CW
  int w = id & 63, n = (id >> 6) & 1023, b = id >> 16;
  float wwn = ww[b * N_ + n];
  float m = mem[id];
  mem[id] = m * (1.0f - wwn * eraseg[b * CW_ + w]) + wwn * wvecg[b * CW_ + w];
  if (w == 0) prec[b * N_ + n] = (1.0f - sumwv[b]) * prec[b * N_ + n] + wwn;
}

// ---------------- read weights + read vectors (per batch) -------------------
__global__ __launch_bounds__(1024) void read_kernel(
    const float* __restrict__ mem, const float* __restrict__ rdkeys,
    const float* __restrict__ rdstr, const float* __restrict__ modes,
    const float* __restrict__ fw, const float* __restrict__ bw,
    float* rw, float* readv) {
  __shared__ float s_red[1024];
  __shared__ float s_rw[4 * 1024];
  int b = blockIdx.x, tid = threadIdx.x, n = tid;
  const float* mrow = mem + ((size_t)(b * N_ + n)) * CW_;
  float dots[4] = {0, 0, 0, 0}, kn[4] = {0, 0, 0, 0}, norm = 0;
  for (int w = 0; w < CW_; ++w) {
    float mv = mrow[w];
    norm += mv * mv;
#pragma unroll
    for (int r = 0; r < 4; ++r) {
      float kv = rdkeys[(b * R_ + r) * CW_ + w];
      dots[r] += mv * kv; kn[r] += kv * kv;
    }
  }
  float inv_norm = 1.0f / (sqrtf(norm) + EPS_);
  for (int r = 0; r < 4; ++r) {
    float score = dots[r] * inv_norm / (sqrtf(kn[r]) + EPS_) * rdstr[b * R_ + r];
    s_red[tid] = score; __syncthreads();
    for (int s = 512; s > 0; s >>= 1) { if (tid < s) s_red[tid] = fmaxf(s_red[tid], s_red[tid + s]); __syncthreads(); }
    float mx = s_red[0]; __syncthreads();
    float e = expf(score - mx);
    s_red[tid] = e; __syncthreads();
    for (int s = 512; s > 0; s >>= 1) { if (tid < s) s_red[tid] += s_red[tid + s]; __syncthreads(); }
    float cw = e / s_red[0];
    __syncthreads();
    float m0 = modes[(b * R_ + r) * 3 + 0];
    float m1 = modes[(b * R_ + r) * 3 + 1];
    float m2 = modes[(b * R_ + r) * 3 + 2];
    float rn = m0 * bw[(b * R_ + r) * N_ + n] + m1 * fw[(b * R_ + r) * N_ + n] + m2 * cw;
    rw[(b * R_ + r) * N_ + n] = rn;
    s_rw[r * N_ + n] = rn;
  }
  __syncthreads();
  // read_vecs[r][w] = sum_n rw[r][n] * mem[n][w]
  int rr = (tid >> 6) & 3, w = tid & 63, part = tid >> 8;
  float acc = 0;
  for (int nn = part; nn < N_; nn += 4)
    acc += s_rw[rr * N_ + nn] * mem[(((size_t)(b * N_ + nn)) << 6) + w];
  s_red[tid] = acc; __syncthreads();
  if (tid < 256)
    readv[b * RV_ + tid] = s_red[tid] + s_red[tid + 256] + s_red[tid + 512] + s_red[tid + 768];
}

// ---------------- output projection -----------------------------------------
__global__ __launch_bounds__(256) void y_kernel(
    const float* __restrict__ outg, const float* __restrict__ readv,
    const float* __restrict__ w_out, const float* __restrict__ b_out,
    float* yout, int t) {
  int b = blockIdx.x, i = threadIdx.x;
  float acc = b_out[i];
  const float* wr = w_out + (size_t)i * (H_ + RV_);
  for (int k = 0; k < H_; ++k) acc += outg[b * H_ + k] * wr[k];
  for (int k = 0; k < RV_; ++k) acc += readv[b * RV_ + k] * wr[H_ + k];
  yout[(b * T_ + t) * IN_ + i] = acc;
}

// ---------------- host side --------------------------------------------------
extern "C" void kernel_launch(void* const* d_in, const int* in_sizes, int n_in,
                              void* d_out, int out_size, void* d_ws, size_t ws_size,
                              hipStream_t stream) {
  (void)in_sizes; (void)n_in; (void)out_size; (void)ws_size;
  const float* x     = (const float*)d_in[0];
  const float* w_ih0 = (const float*)d_in[1];
  const float* w_hh0 = (const float*)d_in[2];
  const float* b_ih0 = (const float*)d_in[3];
  const float* b_hh0 = (const float*)d_in[4];
  const float* w_ih1 = (const float*)d_in[5];
  const float* w_hh1 = (const float*)d_in[6];
  const float* b_ih1 = (const float*)d_in[7];
  const float* b_hh1 = (const float*)d_in[8];
  const float* w_if  = (const float*)d_in[9];
  const float* b_if  = (const float*)d_in[10];
  const float* w_out = (const float*)d_in[11];
  const float* b_out = (const float*)d_in[12];
  const float* h0    = (const float*)d_in[13];

  // ---- workspace carve (~43 MB total; link matrix dominates) ----
  char* ws = (char*)d_ws;
  size_t off = 0;
  auto carve = [&](size_t bytes) -> void* {
    void* p = ws + off;
    off = (off + bytes + 255) & ~(size_t)255;
    return p;
  };
  _Float16* w0h  = (_Float16*)carve((size_t)H4_ * 1024 * 2);          // 4 MB
  _Float16* w1h  = (_Float16*)carve((size_t)H4_ * 1024 * 2);          // 4 MB
  _Float16* wifh = (_Float16*)carve((size_t)30 * 16 * 512 * 2);       // 480 KB
  float* hA   = (float*)carve(B_ * H_ * 4);
  float* cA   = (float*)carve(B_ * H_ * 4);
  float* hB   = (float*)carve(B_ * H_ * 4);
  float* cB   = (float*)carve(B_ * H_ * 4);
  float* outg = (float*)carve(B_ * H_ * 4);
  float* xig  = (float*)carve(B_ * XIP_ * 4);
  float* mem  = (float*)carve((size_t)B_ * N_ * CW_ * 4);             // 2 MB
  float* link = (float*)carve((size_t)B_ * N_ * N_ * 4);              // 32 MB
  // zero-initialized state block (contiguous for one fill)
  const int NZ = B_*N_ + B_*R_*N_ + B_*N_ + B_*N_ + B_*R_*N_ + B_*R_*N_; // 122880
  float* zstate = (float*)carve((size_t)NZ * 4);
  float* prec  = zstate;
  float* rw    = prec  + B_ * N_;
  float* ww    = rw    + B_ * R_ * N_;
  float* usage = ww    + B_ * N_;
  float* fw    = usage + B_ * N_;
  float* bw    = fw    + B_ * R_ * N_;
  float* rdkeys = (float*)carve(B_ * R_ * CW_ * 4);
  float* rdstr  = (float*)carve(B_ * R_ * 4);
  float* eraseg = (float*)carve(B_ * CW_ * 4);
  float* wvecg  = (float*)carve(B_ * CW_ * 4);
  float* modesg = (float*)carve(B_ * R_ * 3 * 4);
  float* sumwv  = (float*)carve(B_ * 4);
  float* readv  = (float*)carve(B_ * RV_ * 4);

  // ---- one-time per call: pack weights to f16 fragments, init state ----
  pack_w_kernel<<<(H4_ * 1024) / 256, 256, 0, stream>>>(w0h, w_ih0, w_hh0, 32, 2048, H4_ * 1024);
  pack_w_kernel<<<(H4_ * 1024) / 256, 256, 0, stream>>>(w1h, w_ih1, w_hh1, 32, 2048, H4_ * 1024);
  pack_w_kernel<<<(30 * 16 * 512) / 256, 256, 0, stream>>>(wifh, w_if, nullptr, 16, 471, 30 * 16 * 512);
  init_h_kernel<<<(B_ * H_ + 255) / 256, 256, 0, stream>>>(h0, hA, cA, hB, cB);
  fill_kernel<<<(B_ * N_ * CW_) / 256, 256, 0, stream>>>(mem, EPS_, B_ * N_ * CW_);
  fill_kernel<<<(B_ * N_ * N_) / 256, 256, 0, stream>>>(link, 0.0f, B_ * N_ * N_);
  fill_kernel<<<(NZ + 255) / 256, 256, 0, stream>>>(zstate, 0.0f, NZ);

  // ---- sequential recurrence over T timesteps ----
  for (int t = 0; t < T_; ++t) {
    lstm_step_kernel<<<1, 256, 0, stream>>>(x, b_ih0, b_hh0, b_ih1, b_hh1, b_if,
                                            w0h, w1h, wifh, hA, cA, hB, cB, outg, xig, t);
    content_kernel<<<B_, 1024, 0, stream>>>(xig, mem, rdkeys, rdstr, eraseg, wvecg,
                                            modesg, ww, rw, usage, sumwv);
    link_fw_kernel<<<B_ * N_, 256, 0, stream>>>(link, ww, prec, rw, fw);
    bw_kernel<<<B_, 1024, 0, stream>>>(link, rw, bw);
    mem_write_kernel<<<(B_ * N_ * CW_) / 256, 256, 0, stream>>>(mem, ww, eraseg, wvecg, prec, sumwv);
    read_kernel<<<B_, 1024, 0, stream>>>(mem, rdkeys, rdstr, modesg, fw, bw, rw, readv);
    y_kernel<<<B_, 256, 0, stream>>>(outg, readv, w_out, b_out, (float*)d_out, t);
  }
}
